// Transformer_3582002724979
// MI455X (gfx1250) — compile-verified, matrix operand order checked by reading, not run on hardware
//
#include <hip/hip_runtime.h>
#include <cstdint>

// ---------------- model dims ----------------
#define TT   1024
#define EE   1024
#define HH   16
#define HDIM 64
#define LLAY 4
#define BBAT 2
#define VV   32000
#define FF   4096
#define NTOK (BBAT * TT)   // 2048 rows

typedef __bf16        bf16x16 __attribute__((ext_vector_type(16)));
typedef __bf16        bf16x2  __attribute__((ext_vector_type(2)));
typedef float         v8f     __attribute__((ext_vector_type(8)));
typedef unsigned int  u32x4   __attribute__((ext_vector_type(4)));
typedef unsigned int  u32x2   __attribute__((ext_vector_type(2)));

union AF { bf16x16 v; u32x4 q[2]; };   // 32 bytes: one 16-bf16 fragment = two 16B loads

// native converts -> v_cvt_(pk_)bf16_f32
__device__ __forceinline__ unsigned short f2bf(float f) {
    return __builtin_bit_cast(unsigned short, (__bf16)f);
}
__device__ __forceinline__ unsigned int f2bf2(float a, float b) {
    bf16x2 h = { (__bf16)a, (__bf16)b };
    return __builtin_bit_cast(unsigned int, h);
}

// CDNA5 16-bit fragment K schedule (ISA 7.12.2): per lane the 16 elements are two
// contiguous K runs:  k = half*8 + {0..7}  and  k = 16 + half*8 + {0..7}.
// A-type fragment from a row-major tile (K contiguous, leading dim ld halves):
__device__ __forceinline__ bf16x16 load_a_rm(const unsigned short* p, int ld, int lane) {
    int row = lane & 15, half = lane >> 4;
    const unsigned short* r = p + row * ld + half * 8;
    AF f;
    f.q[0] = *reinterpret_cast<const u32x4*>(r);       // k: half*8 + 0..7
    f.q[1] = *reinterpret_cast<const u32x4*>(r + 16);  // k: 16 + half*8 + 0..7
    return f.v;
}
// B-type fragment from a fragment-major packed LDS tile: lane-contiguous 16 halves
__device__ __forceinline__ bf16x16 load_b_pk(const unsigned short* p, int lane) {
    const unsigned short* r = p + lane * 16;
    AF f;
    f.q[0] = *reinterpret_cast<const u32x4*>(r);
    f.q[1] = *reinterpret_cast<const u32x4*>(r + 8);
    return f.v;
}
// packed-slot index for element k (0..31) within a lane's 16 halves
__device__ __forceinline__ int pk_slot(int k) { return ((k & 16) >> 1) + (k & 7); }

// ---------------- embedding + sinusoidal positional encoding ----------------
__global__ __launch_bounds__(256) void embed_kernel(const int* __restrict__ tokens,
                                                    const float* __restrict__ emb,
                                                    float* __restrict__ x) {
    int n   = blockIdx.x;            // n = b*T + t
    int t   = n % TT;
    int tok = tokens[n];
    const float c = 2.0f * 9.210340372f / (float)EE;  // 2*ln(10000)/E
    int e = threadIdx.x * 4;                          // 256 threads x 4 = E
    float4 em = *reinterpret_cast<const float4*>(&emb[(size_t)tok * EE + e]);
    float a0 = (float)t * __expf(-c * (float)(e >> 1));
    float a1 = (float)t * __expf(-c * (float)((e >> 1) + 1));
    float4 o;
    o.x = em.x + __sinf(a0);
    o.y = em.y + __cosf(a0);
    o.z = em.z + __sinf(a1);
    o.w = em.w + __cosf(a1);
    *reinterpret_cast<float4*>(&x[(size_t)n * EE + e]) = o;
}

// ---------------- layernorm: f32 out (optional) + bf16 out (optional) ----------------
__global__ __launch_bounds__(256) void layernorm_kernel(const float* __restrict__ x,
                                                        const float* __restrict__ g,
                                                        const float* __restrict__ b,
                                                        float* __restrict__ yF,
                                                        unsigned short* __restrict__ yB) {
    int row = blockIdx.x, tid = threadIdx.x;
    int e = tid * 4;                                   // one float4 per thread
    float4 v4 = *reinterpret_cast<const float4*>(&x[(size_t)row * EE + e]);
    __shared__ float red[256];
    red[tid] = v4.x + v4.y + v4.z + v4.w;
    __syncthreads();
    for (int o = 128; o > 0; o >>= 1) { if (tid < o) red[tid] += red[tid + o]; __syncthreads(); }
    float mu = red[0] / (float)EE;
    __syncthreads();
    float d0 = v4.x - mu, d1 = v4.y - mu, d2 = v4.z - mu, d3 = v4.w - mu;
    red[tid] = d0 * d0 + d1 * d1 + d2 * d2 + d3 * d3;
    __syncthreads();
    for (int o = 128; o > 0; o >>= 1) { if (tid < o) red[tid] += red[tid + o]; __syncthreads(); }
    float inv = rsqrtf(red[0] / (float)EE + 1e-5f);
    float4 g4 = *reinterpret_cast<const float4*>(&g[e]);
    float4 b4 = *reinterpret_cast<const float4*>(&b[e]);
    float o0 = d0 * inv * g4.x + b4.x;
    float o1 = d1 * inv * g4.y + b4.y;
    float o2 = d2 * inv * g4.z + b4.z;
    float o3 = d3 * inv * g4.w + b4.w;
    if (yF) { float4 o4 = {o0, o1, o2, o3};
              *reinterpret_cast<float4*>(&yF[(size_t)row * EE + e]) = o4; }
    if (yB) { u32x2 pk = { f2bf2(o0, o1), f2bf2(o2, o3) };
              *reinterpret_cast<u32x2*>(&yB[(size_t)row * EE + e]) = pk; }
}

// ---------------- bf16 WMMA GEMM: C = act(A_bf16 @ W_f32 + bias) (+resid) ----------------
// Double-buffered pipeline, one barrier per K-step. W staging loads K-row pairs so
// conversion is genuine v_cvt_pk_bf16_f32 -> ds_store_b32 into the fragment-major tile.
__global__ __launch_bounds__(256) void gemm_bf16(const unsigned short* __restrict__ A, // [M,K] bf16
                                                 const float* __restrict__ W,          // [K,N] f32
                                                 const float* __restrict__ bias,       // [N] or null
                                                 const float* __restrict__ resid,      // [M,N] or null
                                                 float* __restrict__ outF,             // [M,N] or null
                                                 unsigned short* __restrict__ outB,    // [M,N] or null
                                                 unsigned short* __restrict__ outBT,   // per-head transposed bf16 or null
                                                 int M, int N, int K, int relu) {
    const int BM = 128, BN = 64, BK = 32;
    __shared__ __align__(16) unsigned short sA[2][BM * BK];       // 2 x 8 KB, row-major
    __shared__ __align__(16) unsigned short sBp[2][4 * 32 * 16];  // 2 x 4 KB, fragment-major
    int tid  = threadIdx.x;
    int wave = tid >> 5, lane = tid & 31;
    int m0 = blockIdx.y * BM;
    int n0 = blockIdx.x * BN;

    int kp = (tid >> 4) << 1;          // even K row owned by this thread (0..30)
    int wc = (tid & 15) << 2;          // N chunk (0..60)

    u32x4  ra[2];
    float4 rw0, rw1;
    auto glob_load = [&](int k0) {
#pragma unroll
        for (int i = 0; i < 2; ++i) {                 // A tile 128x32 bf16: 2 x 16B / thread
            int idx = tid + 256 * i;
            int r = idx >> 2, cc = (idx & 3) << 3;
            ra[i] = *reinterpret_cast<const u32x4*>(&A[(size_t)(m0 + r) * K + k0 + cc]);
        }
        // W tile 32x64 f32: rows (kp, kp+1), same 4 columns -> packable pairs
        rw0 = *reinterpret_cast<const float4*>(&W[(size_t)(k0 + kp)     * N + n0 + wc]);
        rw1 = *reinterpret_cast<const float4*>(&W[(size_t)(k0 + kp + 1) * N + n0 + wc]);
    };
    auto lds_store = [&](int buf) {
#pragma unroll
        for (int i = 0; i < 2; ++i) {
            int idx = tid + 256 * i;
            int r = idx >> 2, cc = (idx & 3) << 3;
            *reinterpret_cast<u32x4*>(&sA[buf][r * BK + cc]) = ra[i];
        }
        int half = (kp >> 3) & 1, slot = pk_slot(kp); // slot(kp+1) == slot+1, same half
        float w0[4] = {rw0.x, rw0.y, rw0.z, rw0.w};
        float w1[4] = {rw1.x, rw1.y, rw1.z, rw1.w};
#pragma unroll
        for (int e = 0; e < 4; ++e) {
            int n = wc + e, j = n >> 4, col = n & 15;
            *reinterpret_cast<unsigned int*>(
                &sBp[buf][(j * 32 + half * 16 + col) * 16 + slot]) = f2bf2(w0[e], w1[e]);
        }
    };

    v8f c[4] = {};
    int nk = K / BK;
    glob_load(0);
    lds_store(0);
    for (int t = 0; t < nk; ++t) {
        __syncthreads();                              // buffer t&1 is ready
        if (t + 1 < nk) glob_load((t + 1) * BK);      // overlap with WMMAs below
        int cur = t & 1;
        bf16x16 af = load_a_rm(&sA[cur][(wave * 16) * BK], BK, lane);
        bf16x16 bfr[4];
#pragma unroll
        for (int j = 0; j < 4; ++j)                   // batch all B-fragment ds loads
            bfr[j] = load_b_pk(&sBp[cur][j * 32 * 16], lane);
#pragma unroll
        for (int j = 0; j < 4; ++j)
            c[j] = __builtin_amdgcn_wmma_f32_16x16x32_bf16(false, af, false, bfr[j],
                                                           (short)0, c[j], false, false);
        if (t + 1 < nk) lds_store(cur ^ 1);
    }
    // epilogue: C row = r + 8*(lane>=16), col = lane&15 (ISA 7.12.2)
    int half = lane >> 4, col = lane & 15;
#pragma unroll
    for (int j = 0; j < 4; ++j) {
#pragma unroll
        for (int r = 0; r < 8; ++r) {
            int gm = m0 + wave * 16 + r + 8 * half;
            int gn = n0 + j * 16 + col;
            float v = c[j][r];
            if (bias)  v += bias[gn];
            if (resid) v += resid[(size_t)gm * N + gn];
            if (relu)  v = fmaxf(v, 0.f);
            if (outF)  outF[(size_t)gm * N + gn] = v;
            if (outB)  outB[(size_t)gm * N + gn] = f2bf(v);
            if (outBT) {            // per-head transposed: Vt[(b*H+h)*HD + d][s]
                int b = gm >> 10, s = gm & (TT - 1);
                int h = gn >> 6,  d = gn & (HDIM - 1);
                outBT[((size_t)(b * HH + h) * HDIM + d) * TT + s] = f2bf(v);
            }
        }
    }
}

// ---------------- flash attention, reference quirk: wei[t,s] = K[t] . Q[s] ----------------
// grid (B*H, T/64); 128 threads = 4 independent waves, each owns 16 "t" rows.
// Q/K fragments: contiguous b128 global loads (batched). V comes pre-transposed per head.
__global__ __launch_bounds__(128) void attn_kernel(const unsigned short* __restrict__ Qb,
                                                   const unsigned short* __restrict__ Kb,
                                                   const unsigned short* __restrict__ Vt,
                                                   unsigned short* __restrict__ Ob) {
    int bh = blockIdx.x;
    int b = bh / HH, h = bh % HH;
    int wave = threadIdx.x >> 5, lane = threadIdx.x & 31;
    int half = lane >> 4, ln = lane & 15;
    int t0 = blockIdx.y * 64 + wave * 16;

    __shared__ __align__(16) unsigned short sP[4][16 * 32]; // per-wave P staging (bf16)
    unsigned short* P = sP[wave];

    size_t base  = ((size_t)b * TT) * EE + (size_t)h * HDIM;      // [n][e] layout
    size_t baseV = (size_t)(b * HH + h) * HDIM * TT;              // [d][s] layout

    // K-rows act as queries: load both d-chunk A-fragments once, reuse for all s
    bf16x16 aK[2];
#pragma unroll
    for (int dc = 0; dc < 2; ++dc) {
        const unsigned short* r = &Kb[base + (size_t)(t0 + ln) * EE + dc * 32 + half * 8];
        AF f;
        f.q[0] = *reinterpret_cast<const u32x4*>(r);
        f.q[1] = *reinterpret_cast<const u32x4*>(r + 16);
        aK[dc] = f.v;
    }

    float m[8], l[8];
    v8f o[4] = {};
#pragma unroll
    for (int r = 0; r < 8; ++r) { m[r] = -INFINITY; l[r] = 0.f; }
    const float scale = 0.125f; // HD^-0.5

    for (int s0 = 0; s0 < t0 + 16; s0 += 32) {   // s0 <= 992, so all s indices < T
        // scores: two 16x16 tiles (cols s0..s0+15, s0+16..s0+31), K-dim = HD via 2 chunks
        AF fq[4];                                 // batch all 4 Q-fragment global loads
#pragma unroll
        for (int j = 0; j < 2; ++j)
#pragma unroll
            for (int dc = 0; dc < 2; ++dc) {
                const unsigned short* r =
                    &Qb[base + (size_t)(s0 + 16 * j + ln) * EE + dc * 32 + half * 8];
                fq[j * 2 + dc].q[0] = *reinterpret_cast<const u32x4*>(r);
                fq[j * 2 + dc].q[1] = *reinterpret_cast<const u32x4*>(r + 16);
            }
        v8f S[2] = {};
#pragma unroll
        for (int j = 0; j < 2; ++j)
#pragma unroll
            for (int dc = 0; dc < 2; ++dc)
                S[j] = __builtin_amdgcn_wmma_f32_16x16x32_bf16(false, aK[dc], false,
                                                               fq[j * 2 + dc].v,
                                                               (short)0, S[j], false, false);
        // causal mask + online softmax (row reductions within 16-lane groups)
#pragma unroll
        for (int r = 0; r < 8; ++r) {
            int t = t0 + r + 8 * half;
            float v0 = S[0][r] * scale; if (s0 + ln > t)      v0 = -INFINITY;
            float v1 = S[1][r] * scale; if (s0 + 16 + ln > t) v1 = -INFINITY;
            float mx = fmaxf(v0, v1);
#pragma unroll
            for (int off = 1; off < 16; off <<= 1) mx = fmaxf(mx, __shfl_xor(mx, off, 32));
            float mn    = fmaxf(m[r], mx);
            float alpha = (mn == -INFINITY) ? 1.0f : __expf(m[r] - mn);
            float p0 = (v0 == -INFINITY) ? 0.f : __expf(v0 - mn);
            float p1 = (v1 == -INFINITY) ? 0.f : __expf(v1 - mn);
            float rs = p0 + p1;
#pragma unroll
            for (int off = 1; off < 16; off <<= 1) rs += __shfl_xor(rs, off, 32);
            l[r] = l[r] * alpha + rs;
            m[r] = mn;
#pragma unroll
            for (int j2 = 0; j2 < 4; ++j2) o[j2][r] *= alpha;
            int row = r + 8 * half;
            P[row * 32 + ln]      = f2bf(p0);   // wave-local LDS; DScnt wait auto-inserted
            P[row * 32 + 16 + ln] = f2bf(p1);
        }
        // O += P(16x32) @ V(32x64); V fragment contiguous in s thanks to [d][s] layout
        AF fv[4];                                 // batch all 4 V-fragment global loads
#pragma unroll
        for (int j2 = 0; j2 < 4; ++j2) {
            const unsigned short* r = &Vt[baseV + (size_t)(j2 * 16 + ln) * TT + s0 + half * 8];
            fv[j2].q[0] = *reinterpret_cast<const u32x4*>(r);
            fv[j2].q[1] = *reinterpret_cast<const u32x4*>(r + 16);
        }
        bf16x16 aP = load_a_rm(P, 32, lane);
#pragma unroll
        for (int j2 = 0; j2 < 4; ++j2)
            o[j2] = __builtin_amdgcn_wmma_f32_16x16x32_bf16(false, aP, false, fv[j2].v,
                                                            (short)0, o[j2], false, false);
    }
    // finalize: divide by l, store bf16 (feeds the Wo GEMM)
    float invl[8];
#pragma unroll
    for (int r = 0; r < 8; ++r) invl[r] = 1.0f / l[r];
#pragma unroll
    for (int j2 = 0; j2 < 4; ++j2) {
#pragma unroll
        for (int r = 0; r < 8; ++r) {
            int t = t0 + r + 8 * half;
            Ob[base + (size_t)t * EE + j2 * 16 + ln] = f2bf(o[j2][r] * invl[r]);
        }
    }
}

// ---------------- orchestration ----------------
extern "C" void kernel_launch(void* const* d_in, const int* in_sizes, int n_in,
                              void* d_out, int out_size, void* d_ws, size_t ws_size,
                              hipStream_t stream) {
    (void)in_sizes; (void)n_in; (void)out_size; (void)ws_size;
    const int*   tokens = (const int*)  d_in[0];
    const float* emb    = (const float*)d_in[1];
    const float* Wq     = (const float*)d_in[2];
    const float* Wk     = (const float*)d_in[3];
    const float* Wv     = (const float*)d_in[4];
    const float* Wo     = (const float*)d_in[5];
    const float* bo     = (const float*)d_in[6];
    const float* ln1g   = (const float*)d_in[7];
    const float* ln1b   = (const float*)d_in[8];
    const float* ln2g   = (const float*)d_in[9];
    const float* ln2b   = (const float*)d_in[10];
    const float* W1     = (const float*)d_in[11];
    const float* b1     = (const float*)d_in[12];
    const float* W2     = (const float*)d_in[13];
    const float* b2     = (const float*)d_in[14];
    const float* lnfg   = (const float*)d_in[15];
    const float* lnfb   = (const float*)d_in[16];
    const float* Wh     = (const float*)d_in[17];
    const float* bh     = (const float*)d_in[18];
    float* out = (float*)d_out;

    // workspace carve-out (~72 MB)
    char* p = (char*)d_ws;
    auto take = [&](size_t bytes) -> void* {
        void* r = (void*)p; p += (bytes + 255) & ~(size_t)255; return r;
    };
    float*          x   = (float*)         take((size_t)NTOK * EE * 4);
    float*          z   = (float*)         take((size_t)NTOK * EE * 4);
    float*          y   = (float*)         take((size_t)NTOK * EE * 4);
    float*          z2  = (float*)         take((size_t)NTOK * EE * 4);
    unsigned short* zb  = (unsigned short*)take((size_t)NTOK * EE * 2);
    unsigned short* z2b = (unsigned short*)take((size_t)NTOK * EE * 2);
    unsigned short* Qb  = (unsigned short*)take((size_t)NTOK * EE * 2);
    unsigned short* Kb  = (unsigned short*)take((size_t)NTOK * EE * 2);
    unsigned short* Vt  = (unsigned short*)take((size_t)NTOK * EE * 2);  // per-head [d][s]
    unsigned short* Ab  = (unsigned short*)take((size_t)NTOK * EE * 2);
    unsigned short* hb  = (unsigned short*)take((size_t)NTOK * FF * 2);

    embed_kernel<<<NTOK, 256, 0, stream>>>(tokens, emb, x);

    dim3 gE(EE / 64, NTOK / 128);   // N=1024 GEMMs
    dim3 gF(FF / 64, NTOK / 128);   // N=4096 GEMM
    for (int i = 0; i < LLAY; ++i) {
        layernorm_kernel<<<NTOK, 256, 0, stream>>>(x, ln1g + (size_t)i * EE, ln1b + (size_t)i * EE, z, zb);
        gemm_bf16<<<gE, 256, 0, stream>>>(zb, Wq + (size_t)i * EE * EE, nullptr, nullptr, nullptr, Qb, nullptr, NTOK, EE, EE, 0);
        gemm_bf16<<<gE, 256, 0, stream>>>(zb, Wk + (size_t)i * EE * EE, nullptr, nullptr, nullptr, Kb, nullptr, NTOK, EE, EE, 0);
        gemm_bf16<<<gE, 256, 0, stream>>>(zb, Wv + (size_t)i * EE * EE, nullptr, nullptr, nullptr, nullptr, Vt, NTOK, EE, EE, 0);
        attn_kernel<<<dim3(BBAT * HH, TT / 64), 128, 0, stream>>>(Qb, Kb, Vt, Ab);
        gemm_bf16<<<gE, 256, 0, stream>>>(Ab, Wo + (size_t)i * EE * EE, bo + (size_t)i * EE, z, y, nullptr, nullptr, NTOK, EE, EE, 0);
        layernorm_kernel<<<NTOK, 256, 0, stream>>>(y, ln2g + (size_t)i * EE, ln2b + (size_t)i * EE, z2, z2b);
        gemm_bf16<<<gF, 256, 0, stream>>>(z2b, W1 + (size_t)i * EE * FF, b1 + (size_t)i * FF, nullptr, nullptr, hb, nullptr, NTOK, FF, EE, 1);
        gemm_bf16<<<gE, 256, 0, stream>>>(hb, W2 + (size_t)i * FF * EE, b2 + (size_t)i * EE, z2, x, nullptr, nullptr, NTOK, EE, FF, 0);
    }
    layernorm_kernel<<<NTOK, 256, 0, stream>>>(x, lnfg, lnfb, nullptr, zb);
    dim3 gH(VV / 64, NTOK / 128);
    gemm_bf16<<<gH, 256, 0, stream>>>(zb, Wh, bh, nullptr, out, nullptr, nullptr, NTOK, VV, EE, 0);
}
